// BilinearDecoder_43215960932828
// MI455X (gfx1250) — matrix operand order repaired and checked by reading, not compiled, not run on hardware
//
#include <hip/hip_runtime.h>

// CDNA5 / gfx1250 WMMA vector types (probe-confirmed builtin signature)
typedef __attribute__((ext_vector_type(16))) _Float16 v16h;
typedef __attribute__((ext_vector_type(8)))  float    v8f;

#define WDIM    512
#define N_EDGES 131072
// M = 10000 = 625 tiles of 16 (exact), N = 512 = 32 tiles of 16, K = 512 = 16 steps of 32
#define KB_STEPS 16

// ---------------------------------------------------------------------------
// Pack W (f32, row-major [k][n]) into f16 B-fragment layout for
// v_wmma_f32_16x16x32_f16.  B k order chosen here (wave32): lane = h*16 + n_low,
// element i (0..15) holds K = kb*32 + 16*h + i for column N = nb*16 + n_low.
// Fragment flat index: nb*512 + kb*32 + lane  (16 halfs = 32 B each).
// ---------------------------------------------------------------------------
__global__ __launch_bounds__(256) void bilinear_pack_w_kernel(
    const float* __restrict__ W, _Float16* __restrict__ wp) {
  int frag = blockIdx.x * blockDim.x + threadIdx.x;   // 0 .. 16383
  int lane = frag & 31;
  int kb   = (frag >> 5) & 15;
  int nb   = frag >> 9;
  int n    = nb * 16 + (lane & 15);
  int h    = lane >> 4;
  int k0   = kb * 32 + h * 16;
  _Float16* dst = wp + ((size_t)frag << 4);
#pragma unroll
  for (int i = 0; i < 16; ++i)
    dst[i] = (_Float16)W[(size_t)(k0 + i) * WDIM + n];
}

// ---------------------------------------------------------------------------
// zW = z @ W  via WMMA f16->f32.  One wave per 16x16 output tile.
// Block = 256 threads = 8 waves; 8 consecutive tiles share the same M-tile
// (8 | 32 N-tiles per M-tile) so A rows hit WGP$/L2 on re-reads.
// A k order chosen here (wave32, 16-bit, 16x32): lane = h*16 + m_low holds row M = m_low,
// element i -> K = kb*32 + 16*h + i: one contiguous 16-float
// run of z, converted f32->f16 in registers.
// ---------------------------------------------------------------------------
__global__ __launch_bounds__(256) void bilinear_gemm_zw_kernel(
    const float* __restrict__ z, const _Float16* __restrict__ wp,
    float* __restrict__ zw) {
  int wave = threadIdx.x >> 5;
  int lane = threadIdx.x & 31;
  int tile = blockIdx.x * 8 + wave;   // 0 .. 19999
  int mt   = tile >> 5;               // 0 .. 624
  int nb   = tile & 31;               // 0 .. 31
  int h    = lane >> 4;
  int ml   = lane & 15;

  const float* zr = z + (size_t)(mt * 16 + ml) * WDIM;

  v8f acc = {};
#pragma unroll
  for (int kb = 0; kb < KB_STEPS; ++kb) {
    int k1 = kb * 32 + 16 * h;                // 16-float run
    const float4* p = (const float4*)(zr + k1);
    float4 f0 = p[0];
    float4 f1 = p[1];
    float4 f2 = p[2];                         // continues at k1 + 8
    float4 f3 = p[3];

    v16h a;
    a[0]  = (_Float16)f0.x; a[1]  = (_Float16)f0.y;
    a[2]  = (_Float16)f0.z; a[3]  = (_Float16)f0.w;
    a[4]  = (_Float16)f1.x; a[5]  = (_Float16)f1.y;
    a[6]  = (_Float16)f1.z; a[7]  = (_Float16)f1.w;
    a[8]  = (_Float16)f2.x; a[9]  = (_Float16)f2.y;
    a[10] = (_Float16)f2.z; a[11] = (_Float16)f2.w;
    a[12] = (_Float16)f3.x; a[13] = (_Float16)f3.y;
    a[14] = (_Float16)f3.z; a[15] = (_Float16)f3.w;

    v16h b = *(const v16h*)(wp + ((size_t)(nb * 512 + kb * 32 + lane) << 4));

    // (neg_a, A, neg_b, B, c_mod, C, reuse_a, reuse_b)
    acc = __builtin_amdgcn_wmma_f32_16x16x32_f16(
        false, a, false, b, (short)0, acc, false, false);
  }

  // C/D layout: VGPR r <-> M = r + 8*h, N = ml
  int m0 = mt * 16 + 8 * h;
  int n  = nb * 16 + ml;
#pragma unroll
  for (int r = 0; r < 8; ++r)
    zw[(size_t)(m0 + r) * WDIM + n] = acc[r];
}

// ---------------------------------------------------------------------------
// Per-edge score: sigmoid( zW[row] . z[col] ).  One wave32 per edge.
// Each lane handles 4 interleaved float4s (coalesced 512 B per access),
// then a 5-step __shfl_xor wave reduction.  Both tables live in L2 (20 MB ea).
// ---------------------------------------------------------------------------
__global__ __launch_bounds__(256) void bilinear_edge_score_kernel(
    const float* __restrict__ zw, const float* __restrict__ z,
    const long long* __restrict__ edges, float* __restrict__ out) {
  int gid  = blockIdx.x * blockDim.x + threadIdx.x;
  int e    = gid >> 5;                 // edge index (exactly 131072 waves)
  int lane = threadIdx.x & 31;

  long long r = edges[e];
  long long c = edges[N_EDGES + e];

  const float4* pa = (const float4*)(zw + (size_t)r * WDIM);
  const float4* pb = (const float4*)(z  + (size_t)c * WDIM);

  float s = 0.0f;
#pragma unroll
  for (int j = 0; j < 4; ++j) {
    float4 a = pa[lane + 32 * j];
    float4 b = pb[lane + 32 * j];
    s = fmaf(a.x, b.x, s);
    s = fmaf(a.y, b.y, s);
    s = fmaf(a.z, b.z, s);
    s = fmaf(a.w, b.w, s);
  }
#pragma unroll
  for (int off = 16; off > 0; off >>= 1)
    s += __shfl_xor(s, off, 32);

  if (lane == 0) out[e] = 1.0f / (1.0f + __expf(-s));
}

// ---------------------------------------------------------------------------
// d_in[0] = z  (f32, 10000*512)
// d_in[1] = batch_edges (int64, 2*131072)
// d_in[2] = W  (f32, 512*512)
// d_out   = scores (f32, 131072)
// d_ws    : [0, 512KB)  packed f16 W fragments
//           [512KB, 512KB + 20.48MB)  zW f32
// ---------------------------------------------------------------------------
extern "C" void kernel_launch(void* const* d_in, const int* in_sizes, int n_in,
                              void* d_out, int out_size, void* d_ws, size_t ws_size,
                              hipStream_t stream) {
  const float*     z     = (const float*)d_in[0];
  const long long* edges = (const long long*)d_in[1];
  const float*     W     = (const float*)d_in[2];
  float*           out   = (float*)d_out;

  _Float16* wp = (_Float16*)d_ws;                          // 512*512*2 = 512 KB
  float*    zw = (float*)((char*)d_ws + (512u * 1024u));   // 10000*512*4 = 20.48 MB

  // 32*16*32 = 16384 B-fragments, one per thread
  bilinear_pack_w_kernel<<<64, 256, 0, stream>>>(W, wp);

  // 625 M-tiles * 32 N-tiles = 20000 tiles, 8 waves (tiles) per block
  bilinear_gemm_zw_kernel<<<2500, 256, 0, stream>>>(z, wp, zw);

  // 131072 edges * 32 lanes / 256 threads = 16384 blocks
  bilinear_edge_score_kernel<<<16384, 256, 0, stream>>>(zw, z, edges, out);
}
